// MultiQueryAttention_70531952935031
// MI455X (gfx1250) — matrix operand order repaired and checked by reading, not compiled, not run on hardware
//
#include <hip/hip_runtime.h>
#include <hip/hip_bf16.h>
#include <math.h>

// ---------------- problem constants ----------------
#define BB 2
#define LL 2048
#define DD 1024
#define HH 16
#define KVH 4
#define HD 64
#define GROUPS (HH / KVH)   // 4

typedef __attribute__((ext_vector_type(16))) __bf16 v16bf;
typedef __attribute__((ext_vector_type(8)))  float  v8f;

// ---- CDNA5 async global->LDS path (guarded; fallback = sync vector copy) ----
#if defined(__gfx1250__) && __has_builtin(__builtin_amdgcn_global_load_async_to_lds_b128)
#define HAS_ASYNC 1
typedef int async_v4i __attribute__((__vector_size__(16)));   // matches builtin param pointee
#define GPTR(p) ((__attribute__((address_space(1))) async_v4i*)(p))
#define LPTR(p) ((__attribute__((address_space(3))) async_v4i*)(p))
#else
#define HAS_ASYNC 0
#endif

// ---------------- helpers ----------------
__device__ __forceinline__ unsigned short f32_to_bf16(float f) {
  unsigned int u = __float_as_uint(f);
  u += 0x7FFFu + ((u >> 16) & 1u);   // round-to-nearest-even
  return (unsigned short)(u >> 16);
}

// 16 contiguous bf16 elements -> v16bf (compiler emits 2x b128 loads)
__device__ __forceinline__ v16bf load16(const unsigned short* p) {
  v16bf r;
  const __bf16* q = reinterpret_cast<const __bf16*>(p);
#pragma unroll
  for (int i = 0; i < 16; ++i) r[i] = q[i];
  return r;
}

// two separate 8-element chunks -> v16bf (A-operand gather)
__device__ __forceinline__ v16bf load8x2(const unsigned short* p0,
                                         const unsigned short* p1) {
  v16bf r;
  const __bf16* a = reinterpret_cast<const __bf16*>(p0);
  const __bf16* b = reinterpret_cast<const __bf16*>(p1);
#pragma unroll
  for (int i = 0; i < 8; ++i) { r[i] = a[i]; r[i + 8] = b[i]; }
  return r;
}

// stage 16 contiguous bf16 (32B) global -> LDS
__device__ __forceinline__ void stage16(const unsigned short* __restrict__ gsrc,
                                        unsigned short* ldst) {
#if HAS_ASYNC
  __builtin_amdgcn_global_load_async_to_lds_b128(GPTR(gsrc),     LPTR(ldst),     0, 0);
  __builtin_amdgcn_global_load_async_to_lds_b128(GPTR(gsrc + 8), LPTR(ldst + 8), 0, 0);
#else
  *reinterpret_cast<uint4*>(ldst)     = *reinterpret_cast<const uint4*>(gsrc);
  *reinterpret_cast<uint4*>(ldst + 8) = *reinterpret_cast<const uint4*>(gsrc + 8);
#endif
}

__device__ __forceinline__ void wait_stage() {
#if HAS_ASYNC
  asm volatile("s_wait_asynccnt 0x0" ::: "memory");
#endif
}

// ---------------- f32 -> bf16 conversion ----------------
__global__ void cvt_bf16_kernel(const float* __restrict__ src,
                                unsigned short* __restrict__ dst, int n) {
  int i = blockIdx.x * blockDim.x + threadIdx.x;
  if (i < n) dst[i] = f32_to_bf16(src[i]);
}

// ---------------- weight pre-pack: W[K][N] f32 -> Wt[N][K] bf16 --------------
__global__ void wtrans_kernel(const float* __restrict__ W,
                              unsigned short* __restrict__ Wt, int K, int N) {
  int idx = blockIdx.x * blockDim.x + threadIdx.x;
  if (idx >= K * N) return;
  const int n = idx % N;
  const int k = idx / N;
  Wt[(size_t)n * K + k] = f32_to_bf16(W[idx]);   // coalesced read, one-time
}

// ---------------- tiled bf16 WMMA GEMM: C[MxN] f32 = A[MxK] * Bt[NxK]^T ------
// Bt is pre-transposed (n-major) so BOTH operand stages are coalesced b128.
// grid = (N/64, M/64), block = 128 (4 waves). Double-buffered LDS pipeline.
__global__ void gemm_bf16_wmma(const unsigned short* __restrict__ A,
                               const unsigned short* __restrict__ Bt,
                               float* __restrict__ C,
                               int M, int N, int K) {
  __shared__ unsigned short As[2][64 * 32];   // [m][k] row-major, 2x4KB
  __shared__ unsigned short Bs[2][64 * 32];   // [n][k] row-major, 2x4KB

  const int tid  = threadIdx.x;
  const int wave = tid >> 5;
  const int lane = tid & 31;
  const int m0 = blockIdx.y * 64;
  const int n0 = blockIdx.x * 64;

  const int row  = tid >> 1;       // staging: 64 rows x (2x16 halves)
  const int half = tid & 1;

  v8f acc[4] = {v8f{}, v8f{}, v8f{}, v8f{}};

  const int mloc = lane & 15;          // A row within wave tile
  const int base = (lane >> 4) * 16;   // A K-half select, contiguous k = 16h + s
  const int koff = (lane >> 4) * 16;   // B K-half select, contiguous k = 16h + s (matches A)

  auto stage = [&](int buf, int k0) {
    stage16(A  + (size_t)(m0 + row) * K + k0 + half * 16,
            &As[buf][row * 32 + half * 16]);
    stage16(Bt + (size_t)(n0 + row) * K + k0 + half * 16,
            &Bs[buf][row * 32 + half * 16]);
  };

  stage(0, 0);
  for (int k0 = 0, it = 0; k0 < K; k0 += 32, ++it) {
    const int cur = it & 1;
    wait_stage();
    __syncthreads();                       // staged tile visible; prev compute done
    if (k0 + 32 < K) stage(cur ^ 1, k0 + 32);   // overlap next-tile streaming

    const unsigned short* arow = &As[cur][(wave * 16 + mloc) * 32];
    v16bf af = load8x2(arow + base, arow + 8 + base);
#pragma unroll
    for (int t = 0; t < 4; ++t) {
      const int n = t * 16 + (lane & 15);
      v16bf bf = load16(&Bs[cur][n * 32 + koff]);
      acc[t] = __builtin_amdgcn_wmma_f32_16x16x32_bf16(
          false, af, false, bf, (short)0, acc[t], false, false);
    }
  }

  // writeback: C layout — vgpr v holds row v + 8*(lane>=16), col = lane%16
#pragma unroll
  for (int t = 0; t < 4; ++t) {
    const int n = n0 + t * 16 + (lane & 15);
#pragma unroll
    for (int v = 0; v < 8; ++v) {
      const int m = m0 + wave * 16 + v + (lane >> 4) * 8;
      C[(size_t)m * N + n] = acc[t][v];
    }
  }
}

// ---------------- RoPE (+optional 1/sqrt(HD) scale), [B,L,NH,HD]f32 -> [B,NH,L,HD]bf16
__global__ void rope_kernel(const float* __restrict__ X,
                            unsigned short* __restrict__ Y,
                            int NH, float scale, int total) {
  int idx = blockIdx.x * blockDim.x + threadIdx.x;
  if (idx >= total) return;
  const int d  = idx & (HD - 1);
  int t        = idx >> 6;         // /HD
  const int nh = t % NH;  t /= NH;
  const int l  = t % LL;
  const int b  = t / LL;

  const float v = X[idx];
  const float other = X[idx + ((d < 32) ? 32 : -32)];
  const float rot = (d < 32) ? -other : other;

  const int dh = d & 31;
  const float inv_freq = __powf(10000.0f, -(float)dh / 32.0f);
  const float ang = (float)l * inv_freq;
  float s, c;
  __sincosf(ang, &s, &c);

  const float out = (v * c + rot * s) * scale;
  Y[(((size_t)b * NH + nh) * LL + l) * HD + d] = f32_to_bf16(out);
}

// ---------------- V transpose: [B,L,KVH,HD]f32 -> [B,KVH,HD,L]bf16 -----------
__global__ void vtrans_kernel(const float* __restrict__ X,
                              unsigned short* __restrict__ Y, int total) {
  int idx = blockIdx.x * blockDim.x + threadIdx.x;
  if (idx >= total) return;
  const int d  = idx & (HD - 1);
  int t        = idx >> 6;
  const int kv = t % KVH;  t /= KVH;
  const int l  = t % LL;
  const int b  = t / LL;
  Y[(((size_t)b * KVH + kv) * HD + d) * LL + l] = f32_to_bf16(X[idx]);
}

// ---------------- flash attention: 1 wave per 16-query tile ------------------
// Q: [B,H,L,HD] bf16 (RoPE'd, pre-scaled by 1/8)
// K: [B,KVH,L,HD] bf16 (RoPE'd)
// Vt:[B,KVH,HD,L] bf16
// O: [B,L,H*HD] bf16
__global__ void attn_kernel(const unsigned short* __restrict__ Qb,
                            const unsigned short* __restrict__ Kb,
                            const unsigned short* __restrict__ Vt,
                            unsigned short* __restrict__ Ob) {
  const int b  = blockIdx.z;
  const int h  = blockIdx.y;
  const int q0 = blockIdx.x * 16;
  const int kvh = h / GROUPS;
  const int lane = threadIdx.x;

  const unsigned short* Qp = Qb + (((size_t)b * HH + h) * LL + q0) * HD;
  const unsigned short* Kp = Kb + (((size_t)b * KVH + kvh) * LL) * HD;
  const unsigned short* Vp = Vt + (((size_t)b * KVH + kvh) * HD) * LL;

  const int mloc = lane & 15;
  const int base = (lane >> 4) * 16;
  const int koff = (lane >> 4) * 16;

  // Q fragments for head-dim chunks [0,32) and [32,64)
  v16bf qf[2];
  qf[0] = load8x2(Qp + mloc * HD + base,      Qp + mloc * HD + 8 + base);
  qf[1] = load8x2(Qp + mloc * HD + 32 + base, Qp + mloc * HD + 40 + base);

  const float slope = exp2f(-0.5f * (float)(h + 1));   // ALiBi, H=16 (pow2)

  float rowM[8], rowL[8];
  v8f acc[4] = {v8f{}, v8f{}, v8f{}, v8f{}};
#pragma unroll
  for (int v = 0; v < 8; ++v) { rowM[v] = -1e30f; rowL[v] = 0.0f; }

  __shared__ unsigned short Ps[16 * 32];   // P tile staged for A-operand reshape

  for (int j = 0; j < LL; j += 32) {
    // prefetch next K/V block into cache while computing this one
    if (j + 32 < LL) {
      __builtin_prefetch(Kp + (size_t)(j + 32 + lane) * HD, 0, 1);
      __builtin_prefetch(Vp + (size_t)lane * LL + j + 32, 0, 1);
      __builtin_prefetch(Vp + (size_t)(lane + 32) * LL + j + 32, 0, 1);
    }

    // ---- S = (Q/8) @ K^T for 32 keys (two 16-key subtiles) + ALiBi ----
    v8f s[2];
#pragma unroll
    for (int sub = 0; sub < 2; ++sub) {
      const int kk = j + sub * 16 + (lane & 15);
      v16bf bk0 = load16(Kp + (size_t)kk * HD + koff);        // dims [0,32)
      v16bf bk1 = load16(Kp + (size_t)kk * HD + 32 + koff);   // dims [32,64)
      v8f sv{};
      sv = __builtin_amdgcn_wmma_f32_16x16x32_bf16(false, qf[0], false, bk0,
                                                   (short)0, sv, false, false);
      sv = __builtin_amdgcn_wmma_f32_16x16x32_bf16(false, qf[1], false, bk1,
                                                   (short)0, sv, false, false);
#pragma unroll
      for (int v = 0; v < 8; ++v) {
        const int q = q0 + v + (lane >> 4) * 8;
        sv[v] += slope * (float)(kk - q);
      }
      s[sub] = sv;
    }

    // ---- online softmax (rows spread across the 16 lanes of each half) ----
    float alpha[8];
#pragma unroll
    for (int v = 0; v < 8; ++v) {
      float mx = fmaxf(s[0][v], s[1][v]);
#pragma unroll
      for (int d = 1; d < 16; d <<= 1) mx = fmaxf(mx, __shfl_xor(mx, d, 32));
      const float mN = fmaxf(rowM[v], mx);
      alpha[v] = __expf(rowM[v] - mN);
      const float p0 = __expf(s[0][v] - mN);
      const float p1 = __expf(s[1][v] - mN);
      s[0][v] = p0; s[1][v] = p1;
      float rs = p0 + p1;
#pragma unroll
      for (int d = 1; d < 16; d <<= 1) rs += __shfl_xor(rs, d, 32);
      rowL[v] = rowL[v] * alpha[v] + rs;
      rowM[v] = mN;
    }

    // ---- P (C-layout) -> LDS row-major [16][32] -> A-operand fragment ----
#pragma unroll
    for (int sub = 0; sub < 2; ++sub)
#pragma unroll
      for (int v = 0; v < 8; ++v)
        Ps[(v + (lane >> 4) * 8) * 32 + sub * 16 + (lane & 15)] =
            f32_to_bf16(s[sub][v]);
    __syncthreads();
    v16bf pf = load8x2(&Ps[mloc * 32 + base], &Ps[mloc * 32 + 8 + base]);

    // rescale running output then accumulate P @ V
#pragma unroll
    for (int t = 0; t < 4; ++t)
#pragma unroll
      for (int v = 0; v < 8; ++v) acc[t][v] *= alpha[v];
#pragma unroll
    for (int t = 0; t < 4; ++t) {
      const int d = t * 16 + (lane & 15);
      v16bf vf = load16(Vp + (size_t)d * LL + j + koff);
      acc[t] = __builtin_amdgcn_wmma_f32_16x16x32_bf16(
          false, pf, false, vf, (short)0, acc[t], false, false);
    }
    __syncthreads();
  }

  // ---- epilogue: normalize and store [B,L,H*HD] ----
#pragma unroll
  for (int t = 0; t < 4; ++t) {
    const int d = t * 16 + (lane & 15);
#pragma unroll
    for (int v = 0; v < 8; ++v) {
      const int q = q0 + v + (lane >> 4) * 8;
      const float o = acc[t][v] / rowL[v];
      Ob[((size_t)b * LL + q) * (HH * HD) + h * HD + d] = f32_to_bf16(o);
    }
  }
}

// ---------------- host orchestration ----------------
extern "C" void kernel_launch(void* const* d_in, const int* in_sizes, int n_in,
                              void* d_out, int out_size, void* d_ws, size_t ws_size,
                              hipStream_t stream) {
  const float* x  = (const float*)d_in[0];
  const float* Wq = (const float*)d_in[1];
  const float* Wk = (const float*)d_in[2];
  const float* Wv = (const float*)d_in[3];
  const float* Wo = (const float*)d_in[4];

  const int Mrows = BB * LL;            // 4096
  const int nX  = BB * LL * DD;         // 4,194,304
  const int nWq = DD * HH * HD;         // 1,048,576
  const int nWk = DD * KVH * HD;        // 262,144
  const int nWo = HH * HD * DD;         // 1,048,576

  char* ws = (char*)d_ws;
  size_t off = 0;
  auto alloc = [&](size_t bytes) { char* p = ws + off; off += (bytes + 255) & ~(size_t)255; return p; };

  unsigned short* xb  = (unsigned short*)alloc((size_t)nX  * 2);
  unsigned short* wqt = (unsigned short*)alloc((size_t)nWq * 2);  // [1024][1024]
  unsigned short* wkt = (unsigned short*)alloc((size_t)nWk * 2);  // [256][1024]
  unsigned short* wvt = (unsigned short*)alloc((size_t)nWk * 2);  // [256][1024]
  unsigned short* wot = (unsigned short*)alloc((size_t)nWo * 2);  // [1024][1024]
  float*          Qf  = (float*)alloc((size_t)Mrows * (HH * HD) * 4);
  float*          Kf  = (float*)alloc((size_t)Mrows * (KVH * HD) * 4);
  float*          Vf  = (float*)alloc((size_t)Mrows * (KVH * HD) * 4);
  unsigned short* qb  = (unsigned short*)alloc((size_t)Mrows * (HH * HD) * 2);
  unsigned short* kb  = (unsigned short*)alloc((size_t)Mrows * (KVH * HD) * 2);
  unsigned short* vtb = (unsigned short*)alloc((size_t)Mrows * (KVH * HD) * 2);
  unsigned short* ob  = (unsigned short*)alloc((size_t)Mrows * (HH * HD) * 2);

  const int CT = 256;
  cvt_bf16_kernel<<<(nX + CT - 1) / CT, CT, 0, stream>>>(x, xb, nX);
  wtrans_kernel<<<(nWq + CT - 1) / CT, CT, 0, stream>>>(Wq, wqt, DD, HH * HD);
  wtrans_kernel<<<(nWk + CT - 1) / CT, CT, 0, stream>>>(Wk, wkt, DD, KVH * HD);
  wtrans_kernel<<<(nWk + CT - 1) / CT, CT, 0, stream>>>(Wv, wvt, DD, KVH * HD);
  wtrans_kernel<<<(nWo + CT - 1) / CT, CT, 0, stream>>>(Wo, wot, HH * HD, DD);

  // projections: Q = x@Wq (4096x1024x1024), K/V = x@Wk/Wv (4096x256x1024)
  gemm_bf16_wmma<<<dim3((HH * HD) / 64, Mrows / 64), 128, 0, stream>>>(
      xb, wqt, Qf, Mrows, HH * HD, DD);
  gemm_bf16_wmma<<<dim3((KVH * HD) / 64, Mrows / 64), 128, 0, stream>>>(
      xb, wkt, Kf, Mrows, KVH * HD, DD);
  gemm_bf16_wmma<<<dim3((KVH * HD) / 64, Mrows / 64), 128, 0, stream>>>(
      xb, wvt, Vf, Mrows, KVH * HD, DD);

  // RoPE + layout changes (Q gets 1/sqrt(HD) folded in)
  const int nQ = Mrows * HH * HD, nK = Mrows * KVH * HD;
  rope_kernel<<<(nQ + CT - 1) / CT, CT, 0, stream>>>(Qf, qb, HH, 0.125f, nQ);
  rope_kernel<<<(nK + CT - 1) / CT, CT, 0, stream>>>(Kf, kb, KVH, 1.0f, nK);
  vtrans_kernel<<<(nK + CT - 1) / CT, CT, 0, stream>>>(Vf, vtb, nK);

  // flash attention: one wave per 16-query tile
  attn_kernel<<<dim3(LL / 16, HH, BB), 32, 0, stream>>>(qb, kb, vtb, ob);

  // output projection -> f32 result
  gemm_bf16_wmma<<<dim3(DD / 64, Mrows / 64), 128, 0, stream>>>(
      ob, wot, (float*)d_out, Mrows, DD, HH * HD);
}
